// ContrastiveLossLabelled_43619687858278
// MI455X (gfx1250) — matrix operand order repaired and checked by reading, not compiled, not run on hardware
//
#include <hip/hip_runtime.h>
#include <hip/hip_fp16.h>
#include <stdint.h>

typedef _Float16 v8h  __attribute__((ext_vector_type(8)));
typedef _Float16 v16h __attribute__((ext_vector_type(16)));
typedef float    v8f  __attribute__((ext_vector_type(8)));

#define B_ROWS 32768
#define D_DIM  768
#define C_RAW  200
#define C_PAD  224              // 14 * 16 (two groups of 7 N tiles)
#define NTG    7                // N tiles per wave group
#define ROWS_PER_BLOCK 128      // 4 M-groups * 32 rows
#define KC     64               // K chunk staged in LDS
#define KSTRIDE (KC + 8)        // padded LDS row stride in halves (144 B)
#define NCHUNK (D_DIM / KC)     // 12

// sched_group_barrier masks
#define SGB_WMMA 0x008          // MFMA/WMMA instructions
#define SGB_DSRD 0x100          // DS read instructions

// ---------------------------------------------------------------------------
// Kernel 1: L2-normalize centers (fp32), write f16, pad rows 200..223 with 0
// ---------------------------------------------------------------------------
__global__ __launch_bounds__(256)
void k_normalize_centers(const float* __restrict__ centers,
                         _Float16* __restrict__ cnH) {
    __shared__ float red[256];
    const int c = blockIdx.x;
    const int t = threadIdx.x;
    if (c < C_RAW) {
        float s = 0.f;
        for (int i = t; i < D_DIM; i += 256) {
            float x = centers[(size_t)c * D_DIM + i];
            s += x * x;
        }
        red[t] = s;
        __syncthreads();
        for (int off = 128; off > 0; off >>= 1) {
            if (t < off) red[t] += red[t + off];
            __syncthreads();
        }
        const float inv = 1.0f / fmaxf(sqrtf(red[0]), 1e-8f);  // EPS_COS
        for (int i = t; i < D_DIM; i += 256)
            cnH[(size_t)c * D_DIM + i] =
                (_Float16)(centers[(size_t)c * D_DIM + i] * inv);
    } else {
        for (int i = t; i < D_DIM; i += 256)
            cnH[(size_t)c * D_DIM + i] = (_Float16)0.0f;
    }
}

// A fragment (ISA 16-bit A 16x32): lane holds row (L&15); halves 0..7 at
// K = ko + 8*(L>>4), halves 8..15 at K = ko + 16 + 8*(L>>4).
#define LOADA(m, ko)                                                           \
    ({                                                                         \
        const _Float16* _ap =                                                  \
            &fT[(mrow + (m) * 16 + lo) * KSTRIDE + (ko) + 8 * hi];             \
        v8h _l = *(const v8h*)_ap;                                             \
        v8h _h = *(const v8h*)(_ap + 16);                                      \
        __builtin_shufflevector(_l, _h, 0, 1, 2, 3, 4, 5, 6, 7, 8, 9, 10, 11,  \
                                12, 13, 14, 15);                               \
    })

// B fragment (32x16 f16): lane holds col (L&15); halves 0..7 at K = ko + 8*(L>>4),
// halves 8..15 at K = ko + 16 + 8*(L>>4) of center row (nbase+n)*16 + (L&15).
#define LOADB(n, ko)                                                           \
    ({                                                                         \
        const _Float16* _bp =                                                  \
            &cT[((nbase + (n)) * 16 + lo) * KSTRIDE + (ko) + 8 * hi];          \
        v8h _blo = *(const v8h*)_bp;                                           \
        v8h _bhi = *(const v8h*)(_bp + 16);                                    \
        __builtin_shufflevector(_blo, _bhi, 0, 1, 2, 3, 4, 5, 6, 7, 8, 9, 10,  \
                                11, 12, 13, 14, 15);                           \
    })

// ---------------------------------------------------------------------------
// Kernel 2: WMMA GEMM (f16 in, f32 acc) + fused |.| row reductions.
//   Block = 128 rows x 224 cols. Wave (g = w>>2, wi = w&3): rows
//   [32*wi, 32*wi+32) x N tiles [7g, 7g+7). Each B fragment feeds 2 WMMAs.
//   sched_group_barrier enforces a depth-2 DS->WMMA software pipeline.
// ---------------------------------------------------------------------------
__global__ __launch_bounds__(256)
void k_cos_wmma(const float* __restrict__ feat,
                const _Float16* __restrict__ cnH,
                const int* __restrict__ labels,
                const unsigned char* __restrict__ labelled,
                float* __restrict__ partials) {
    __shared__ _Float16 fT[ROWS_PER_BLOCK * KSTRIDE];  // 18,432 B
    __shared__ _Float16 cT[C_PAD * KSTRIDE];           // 32,256 B
    __shared__ float    rowS[2][ROWS_PER_BLOCK];       // per-N-group row sums
    __shared__ float    rowT[2][ROWS_PER_BLOCK];
    __shared__ float    red[ROWS_PER_BLOCK];

    const int t  = threadIdx.x;
    const int w  = t >> 5;        // wave id 0..7
    const int L  = t & 31;        // lane
    const int lo = L & 15;
    const int hi = L >> 4;
    const int g  = w >> 2;        // N group 0..1
    const int wi = w & 3;         // M group 0..3
    const int nbase = g * NTG;    // first N tile of this wave
    const int mrow  = wi * 32;    // first local row of this wave
    const int rowBase = blockIdx.x * ROWS_PER_BLOCK;

    const v8f zero = {0.f, 0.f, 0.f, 0.f, 0.f, 0.f, 0.f, 0.f};
    v8f acc[2][NTG];
#pragma unroll
    for (int m = 0; m < 2; ++m)
#pragma unroll
        for (int n = 0; n < NTG; ++n) acc[m][n] = zero;

    // staging thread mapping
    const int fr0 = t >> 4, fk = (t & 15) * 4;  // features: 16 thr/row, 4 f32
    const int cr0 = t >> 3, ch = (t & 7) * 8;   // centers : 8 thr/row, 8 f16

    for (int kc = 0; kc < NCHUNK; ++kc) {
        const int kb = kc * KC;
        // --- stage features fp32 -> f16 (coalesced 16B global loads) ---
#pragma unroll
        for (int rr = 0; rr < 8; ++rr) {
            const int r = rr * 16 + fr0;
            float4 v = *(const float4*)(feat + (size_t)(rowBase + r) * D_DIM + kb + fk);
            union { _Float16 h[4]; uint64_t u; } p;
            p.h[0] = (_Float16)v.x; p.h[1] = (_Float16)v.y;
            p.h[2] = (_Float16)v.z; p.h[3] = (_Float16)v.w;
            *(uint64_t*)&fT[r * KSTRIDE + fk] = p.u;
        }
        // --- stage centers f16 (L2-resident) : 224 rows x 64 halves ---
#pragma unroll
        for (int cc = 0; cc < 7; ++cc) {
            const int c = cc * 32 + cr0;
            uint4 v = *(const uint4*)(cnH + (size_t)c * D_DIM + kb + ch);
            *(uint4*)&cT[c * KSTRIDE + ch] = v;
        }
        __syncthreads();

        // --- prefetch next chunk's feature lines (global_prefetch_b8) ---
        if (kc + 1 < NCHUNK) {
#pragma unroll
            for (int rr = 0; rr < 8; ++rr) {
                const float* pf =
                    feat + (size_t)(rowBase + rr * 16 + fr0) * D_DIM + kb + KC + fk;
                __builtin_prefetch(pf, 0, 1);
            }
        }

        // --- compute: 2 k-steps; each B fragment feeds 2 WMMAs ---
#pragma unroll
        for (int ks = 0; ks < 2; ++ks) {
            const int ko = ks * 32;
            v16h a0 = LOADA(0, ko);
            v16h a1 = LOADA(1, ko);
            v16h bb[2];
            bb[0] = LOADB(0, ko);
            bb[1] = LOADB(1, ko);
#pragma unroll
            for (int n = 0; n < NTG; ++n) {
                v16h bcur = bb[n & 1];
                if (n + 2 < NTG) bb[n & 1] = LOADB(n + 2, ko);
                acc[0][n] = __builtin_amdgcn_wmma_f32_16x16x32_f16(
                    false, a0, false, bcur, (short)0, acc[0][n], false, false);
                acc[1][n] = __builtin_amdgcn_wmma_f32_16x16x32_f16(
                    false, a1, false, bcur, (short)0, acc[1][n], false, false);
            }
#if __has_builtin(__builtin_amdgcn_sched_group_barrier)
            // Enforce the depth-2 pipeline per k-step:
            //   DS(8): 4 A-loads + B0 + B1
            //   5 x [ WMMA(2) ; DS(2): B(n+2) ]
            //   WMMA(4): drain
            __builtin_amdgcn_sched_group_barrier(SGB_DSRD, 8, 0);
#pragma unroll
            for (int it = 0; it < 5; ++it) {
                __builtin_amdgcn_sched_group_barrier(SGB_WMMA, 2, 0);
                __builtin_amdgcn_sched_group_barrier(SGB_DSRD, 2, 0);
            }
            __builtin_amdgcn_sched_group_barrier(SGB_WMMA, 4, 0);
#endif
        }
        __syncthreads();
    }

    // ---- epilogue: per-row S = sum|dot|, T = |dot at label column| ----
    // C/D layout: lane L holds col N = (L&15)+16*(nbase+n), VGPR j -> local
    // row mrow + m*16 + 8*(L>>4) + j.
    int rowg[2][8], lab[2][8];
#pragma unroll
    for (int m = 0; m < 2; ++m)
#pragma unroll
        for (int j = 0; j < 8; ++j) {
            rowg[m][j] = rowBase + mrow + m * 16 + hi * 8 + j;
            lab[m][j]  = labels[rowg[m][j]];
        }
    float S[2][8], T[2][8];
#pragma unroll
    for (int m = 0; m < 2; ++m)
#pragma unroll
        for (int j = 0; j < 8; ++j) { S[m][j] = 0.f; T[m][j] = 0.f; }
#pragma unroll
    for (int n = 0; n < NTG; ++n) {
        const int col = (nbase + n) * 16 + lo;
#pragma unroll
        for (int m = 0; m < 2; ++m)
#pragma unroll
            for (int j = 0; j < 8; ++j) {
                float v = fabsf(acc[m][n][j]);
                S[m][j] += v;
                T[m][j] += (lab[m][j] == col) ? v : 0.0f;
            }
    }
    // butterfly across the 16-lane half (masks <=8 never cross halves)
#pragma unroll
    for (int msk = 1; msk <= 8; msk <<= 1) {
#pragma unroll
        for (int m = 0; m < 2; ++m)
#pragma unroll
            for (int j = 0; j < 8; ++j) {
                S[m][j] += __shfl_xor(S[m][j], msk, 32);
                T[m][j] += __shfl_xor(T[m][j], msk, 32);
            }
    }
    if (lo == 0) {
#pragma unroll
        for (int m = 0; m < 2; ++m)
#pragma unroll
            for (int j = 0; j < 8; ++j) {
                const int lr = mrow + m * 16 + hi * 8 + j;
                rowS[g][lr] = S[m][j];
                rowT[g][lr] = T[m][j];
            }
    }
    __syncthreads();
    // combine the two N groups per row, then fixed-order block reduction
    if (t < ROWS_PER_BLOCK) {
        const float Ssum = rowS[0][t] + rowS[1][t];
        const float Tsum = rowT[0][t] + rowT[1][t];
        red[t] = labelled[rowBase + t]
                     ? (1.0f - 2.0f * Tsum / fmaxf(Ssum, 1e-20f))
                     : 0.0f;
    }
    __syncthreads();
    for (int off = 64; off > 0; off >>= 1) {
        if (t < off) red[t] += red[t + off];
        __syncthreads();
    }
    if (t == 0) partials[blockIdx.x] = red[0];
}

// ---------------------------------------------------------------------------
// Kernel 3: deterministic fixed-order reduction of 256 block partials
// ---------------------------------------------------------------------------
__global__ __launch_bounds__(256)
void k_final_reduce(const float* __restrict__ p, float* __restrict__ out) {
    __shared__ float red[256];
    const int t = threadIdx.x;
    red[t] = p[t];
    __syncthreads();
    for (int off = 128; off > 0; off >>= 1) {
        if (t < off) red[t] += red[t + off];
        __syncthreads();
    }
    if (t == 0) out[0] = red[0];
}

extern "C" void kernel_launch(void* const* d_in, const int* in_sizes, int n_in,
                              void* d_out, int out_size, void* d_ws, size_t ws_size,
                              hipStream_t stream) {
    const float*         feat     = (const float*)d_in[0];
    const float*         centers  = (const float*)d_in[1];
    const int*           labels   = (const int*)d_in[2];
    const unsigned char* labelled = (const unsigned char*)d_in[3];
    float* out = (float*)d_out;

    // workspace layout: [0,1024) block partials, [1024, ...) padded f16 centers
    float*    partials = (float*)d_ws;
    _Float16* cnH      = (_Float16*)((char*)d_ws + 1024);

    k_normalize_centers<<<C_PAD, 256, 0, stream>>>(centers, cnH);
    k_cos_wmma<<<B_ROWS / ROWS_PER_BLOCK, 256, 0, stream>>>(
        feat, cnH, labels, labelled, partials);
    k_final_reduce<<<1, 256, 0, stream>>>(partials, out);
}
